// ChannelModel_19387482374790
// MI455X (gfx1250) — matrix operand order repaired and checked by reading, not compiled, not run on hardware
//
#include <hip/hip_runtime.h>

#define HID   10
#define NL    5
#define TT    512
#define NB    4096
#define GATES 40

typedef __attribute__((ext_vector_type(16))) _Float16     v16h_t;
typedef __attribute__((ext_vector_type(8)))  float        v8f_t;
typedef __attribute__((ext_vector_type(4)))  unsigned int v4u_t;

__device__ __forceinline__ float fast_sigmoid(float x) {
    return __builtin_amdgcn_rcpf(1.f + __expf(-x));   // v_exp_f32 + v_rcp_f32
}
__device__ __forceinline__ float fast_tanh(float x) {
#if __has_builtin(__builtin_amdgcn_tanhf)
    return __builtin_amdgcn_tanhf(x);                 // gfx1250 v_tanh_f32
#else
    return 1.f - 2.f * __builtin_amdgcn_rcpf(__expf(2.f * x) + 1.f);
#endif
}

__global__ __launch_bounds__(32) void lstm5_wmma_kernel(
    const float* __restrict__ x,     // [B, T, 2]
    const float* __restrict__ h0,    // [5, B, 10]
    const float* __restrict__ c0,    // [5, B, 10]
    const float* __restrict__ Wih0,  // [40, 2]
    const float* __restrict__ Wihr,  // [4, 40, 10]
    const float* __restrict__ Whh,   // [5, 40, 10]
    const float* __restrict__ bias,  // [5, 40]
    const float* __restrict__ Wf,    // [1, 10]
    const float* __restrict__ bf,    // [1]
    float* __restrict__ out)         // [B*T] ++ [5*B*10] (h) ++ [5*B*10] (c)
{
    // A-operand staging in WMMA 16-bit A 16x32 layout order, pre-zero-padded.
    __shared__ __align__(16) _Float16 ubuf[NL][16][32];   // 5 KB
    __shared__ float cbuf[NL][16][HID];
    __shared__ float hbuf[NL][16][HID];                   // f32 h for head + hT out
    __shared__ float gbuf[16][48];                        // gate scratch, padded

    const int  lane = threadIdx.x;        // 0..31 (wave32)
    const int  base = blockIdx.x * 16;    // batch base for this wave
    const int  m    = lane & 15;          // matrix row / column-in-tile
    const bool hi   = lane >= 16;

    // ---- zero ubuf (covers the permanently-zero K slots) ----
    for (int idx = lane; idx < NL * 16 * 16; idx += 32)   // 1280 dwords
        ((unsigned int*)ubuf)[idx] = 0u;

    // ---- init h, c state from inputs ----
    for (int idx = lane; idx < NL * 16 * HID; idx += 32) {
        int l  = idx / (16 * HID);
        int r  = idx % (16 * HID);
        int mm = r / HID;
        int hh = r % HID;
        const float hv = h0[(size_t)l * NB * HID + (size_t)(base + mm) * HID + hh];
        const float cv = c0[(size_t)l * NB * HID + (size_t)(base + mm) * HID + hh];
        hbuf[l][mm][hh] = hv;
        cbuf[l][mm][hh] = cv;
        ubuf[l][mm][hh] = (_Float16)hv;   // in-order with the zeroing stores (same wave)
    }
    __syncthreads();

    // ---- precompute B fragments (16-bit B 32x16 layout) + per-column bias ----
    // lane<16: column n=m, K=0..7,16..23 ; lane>=16: column n=m, K=8..15,24..31
    v16h_t bfr[NL][3];
    float  breg[NL][3];
    const int kb = hi ? 8 : 0;
    #pragma unroll
    for (int l = 0; l < NL; ++l) {
        const int kin = (l == 0) ? 2 : HID;
        #pragma unroll
        for (int j = 0; j < 3; ++j) {
            const int col = j * 16 + m;
            breg[l][j] = (col < GATES) ? bias[l * GATES + col] : 0.f;
            v16h_t f;
            #pragma unroll
            for (int e = 0; e < 16; ++e) {
                const int k = kb + e + (e >= 8 ? 8 : 0);
                float w = 0.f;
                if (col < GATES) {
                    if (k < HID) {
                        w = Whh[(size_t)l * GATES * HID + col * HID + k];      // h @ W_hh.T
                    } else if (k - HID < kin) {
                        w = (l == 0)
                              ? Wih0[col * 2 + (k - HID)]                       // x @ W_ih0.T
                              : Wihr[(size_t)(l - 1) * GATES * HID + col * HID + (k - HID)];
                    }
                }
                f[e] = (_Float16)w;
            }
            bfr[l][j] = f;
        }
    }

    // final head weights
    float wf[HID];
    #pragma unroll
    for (int hh = 0; hh < HID; ++hh) wf[hh] = Wf[hh];
    const float bfv = bf[0];

    // activation work split: 160 (row, hidden) cells -> 5 per lane
    int am[5], ah[5];
    #pragma unroll
    for (int q = 0; q < 5; ++q) {
        const int idx = lane * 5 + q;
        am[q] = idx / HID;
        ah[q] = idx % HID;
    }

    // software-pipelined per-step x load: lane supplies x[base+m][t][lane>>4]
    const int    xe   = lane >> 4;                      // 0 or 1
    const float* xrow = x + (size_t)(base + m) * TT * 2;
    float xv = xrow[0 * 2 + xe];

    // ---- sequential time loop; layers interleaved per step ----
    for (int t = 0; t < TT; ++t) {
        // stage x_t into ubuf[0] K-slots 10,11 (consumed by hi lanes' A load)
        ubuf[0][m][HID + xe] = (_Float16)xv;
        __syncthreads();

        // issue next step's x load now; overlaps the whole 5-layer step
        const int tn = (t + 1 < TT) ? (t + 1) : t;
        const float xv_next = xrow[tn * 2 + xe];

        #pragma unroll
        for (int l = 0; l < NL; ++l) {
            // A fragment = two ds_load_b128 in exact WMMA layout order:
            // lo lane: K 0..7 @ bytes [0,16), K 16..23 @ [32,48)
            // hi lane: K 8..15 @ [16,32),    K 24..31 @ [48,64)
            union { v16h_t h; v4u_t q[2]; } au;
            const _Float16* up  = &ubuf[l][m][0];
            const int       off = hi ? 8 : 0;
            au.q[0] = *(const v4u_t*)(up + off);
            au.q[1] = *(const v4u_t*)(up + 16 + off);

            // 3 WMMAs cover gate columns 0..47 (40 real); bias rides in C
            #pragma unroll
            for (int j = 0; j < 3; ++j) {
                v8f_t cacc;
                #pragma unroll
                for (int r = 0; r < 8; ++r) cacc[r] = breg[l][j];
                cacc = __builtin_amdgcn_wmma_f32_16x16x32_f16(
                    false, au.h, false, bfr[l][j], (short)0, cacc, false, false);
                const int col = j * 16 + m;
                #pragma unroll
                for (int r = 0; r < 8; ++r)            // unconditional: gbuf padded to 48
                    gbuf[hi ? (r + 8) : r][col] = cacc[r];
            }
            __syncthreads();

            // Activations + state update (torch gate order i, f, g, o)
            #pragma unroll
            for (int q = 0; q < 5; ++q) {
                const int mm = am[q], hh = ah[q];
                const float gi = gbuf[mm][hh];
                const float gf = gbuf[mm][HID + hh];
                const float gg = gbuf[mm][2 * HID + hh];
                const float go = gbuf[mm][3 * HID + hh];
                const float cn = fast_sigmoid(gf) * cbuf[l][mm][hh]
                               + fast_sigmoid(gi) * fast_tanh(gg);
                cbuf[l][mm][hh] = cn;
                const float hn = fast_sigmoid(go) * fast_tanh(cn);
                hbuf[l][mm][hh] = hn;
                const _Float16 hx = (_Float16)hn;
                ubuf[l][mm][hh] = hx;                       // own recurrence input
                if (l < NL - 1) ubuf[l + 1][mm][HID + hh] = hx;  // next layer's input
            }
            __syncthreads();
        }

        // Final 10->1 head on layer-4 output, lanes 0..15 (one per batch row)
        if (!hi) {
            float acc = bfv;
            #pragma unroll
            for (int hh = 0; hh < HID; ++hh) acc += hbuf[NL - 1][m][hh] * wf[hh];
            out[(size_t)(base + m) * TT + t] = fast_sigmoid(acc);
        }
        xv = xv_next;
    }
    __syncthreads();

    // ---- emit final (hT, cT) stacks ----
    const size_t BT = (size_t)NB * TT;
    const size_t CS = (size_t)NL * NB * HID;
    for (int idx = lane; idx < NL * 16 * HID; idx += 32) {
        int l  = idx / (16 * HID);
        int r  = idx % (16 * HID);
        int mm = r / HID;
        int hh = r % HID;
        const size_t off = (size_t)l * NB * HID + (size_t)(base + mm) * HID + hh;
        out[BT + off]      = hbuf[l][mm][hh];
        out[BT + CS + off] = cbuf[l][mm][hh];
    }
}

extern "C" void kernel_launch(void* const* d_in, const int* in_sizes, int n_in,
                              void* d_out, int out_size, void* d_ws, size_t ws_size,
                              hipStream_t stream) {
    (void)in_sizes; (void)n_in; (void)out_size; (void)d_ws; (void)ws_size;
    lstm5_wmma_kernel<<<NB / 16, 32, 0, stream>>>(
        (const float*)d_in[0],   // x
        (const float*)d_in[1],   // h
        (const float*)d_in[2],   // c
        (const float*)d_in[3],   // W_ih0
        (const float*)d_in[4],   // W_ih_rest
        (const float*)d_in[5],   // W_hh
        (const float*)d_in[6],   // b
        (const float*)d_in[7],   // Wf
        (const float*)d_in[8],   // bf
        (float*)d_out);
}
